// SlotAttentionGumbelV1_38766374813835
// MI455X (gfx1250) — matrix operand order repaired and checked by reading, not compiled, not run on hardware
//
#include <hip/hip_runtime.h>
#include <hip/hip_bf16.h>
#include <cstdint>

#define B_    32
#define N_    4096
#define K_    11
#define KP    16       // slots padded to one WMMA M-tile
#define FD_   256
#define D_    256
#define G3D_  768      // 3*D for GRU gates
#define ITERS_ 3
#define EPS_   1e-8f
#define LN_EPS_ 1e-5f
#define SCALE_ 0.0625f // 256^-0.5

typedef __attribute__((ext_vector_type(16))) __bf16 v16bf;
typedef __attribute__((ext_vector_type(8)))  float  v8f;

union BF16Frag { v16bf v; unsigned short h[16]; uint4 q[2]; };

__device__ __forceinline__ unsigned short f2bf(float f) {
  union { float f; uint32_t u; } x; x.f = f;
  uint32_t r = x.u + 0x7FFFu + ((x.u >> 16) & 1u);   // round-to-nearest-even
  return (unsigned short)(r >> 16);
}

__device__ __forceinline__ v8f wmma_bf16(v16bf a, v16bf b, v8f c) {
  return __builtin_amdgcn_wmma_f32_16x16x32_bf16(false, a, false, b, (short)0, c, false, false);
}

// A-fragment: 16(M) x 32(K) bf16, row-major source with leading dim ld.
// Layout (ISA 7.12.2): lanes 0-15: K=k0+0..7 then k0+16..23; lanes 16-31: +8.
__device__ __forceinline__ v16bf load_a_bf16(const unsigned short* base, int ld, int k0, int lane) {
  int m = lane & 15, half = lane >> 4;
  const unsigned short* p = base + (size_t)m * ld + k0 + 8 * half;
  BF16Frag f;
  f.q[0] = *(const uint4*)p;
  f.q[1] = *(const uint4*)(p + 16);
  return f.v;
}

// B-fragment: 32(K) x 16(N) bf16, source stored K-contiguous per column: base[c*ldc + k].
// Layout: lanes 0-15 hold K=k0+0..7 then k0+16..23; lanes 16-31: +8, col = lane&15.
__device__ __forceinline__ v16bf load_b_bf16(const unsigned short* base, int ldc, int k0, int lane) {
  int c = lane & 15, half = lane >> 4;
  const unsigned short* p = base + (size_t)c * ldc + k0 + 8 * half;
  BF16Frag f;
  f.q[0] = *(const uint4*)p;
  f.q[1] = *(const uint4*)(p + 16);
  return f.v;
}

// A-fragment built from fp32 with (x+add)*mul transform (attn renorm / GRU inputs).
__device__ __forceinline__ v16bf load_a_f32(const float* base, int ld, int k0, int lane,
                                            float add, float mul, bool zero) {
  int m = lane & 15, half = lane >> 4;
  const float* p = base + (size_t)m * ld + k0 + 8 * half;
  BF16Frag f;
  if (zero) { f.q[0] = make_uint4(0,0,0,0); f.q[1] = make_uint4(0,0,0,0); return f.v; }
  #pragma unroll
  for (int i = 0; i < 8; i++) f.h[i]     = f2bf((p[i]      + add) * mul);
  #pragma unroll
  for (int i = 0; i < 8; i++) f.h[8 + i] = f2bf((p[16 + i] + add) * mul);
  return f.v;
}

__device__ __forceinline__ v8f zero8() {
  v8f z = {0.f,0.f,0.f,0.f,0.f,0.f,0.f,0.f};
  return z;
}

// ---------------- weight prep: bf16 convert (+transpose for B-fragment layout) --------
__global__ void prep_weights_kernel(const float* __restrict__ wq, const float* __restrict__ wk,
                                    const float* __restrict__ wv, const float* __restrict__ wih,
                                    const float* __restrict__ whh,
                                    unsigned short* __restrict__ wqT, unsigned short* __restrict__ wkT,
                                    unsigned short* __restrict__ wvT, unsigned short* __restrict__ wihB,
                                    unsigned short* __restrict__ whhB) {
  int i = blockIdx.x * blockDim.x + threadIdx.x;
  if (i < D_ * D_) {       // transposed: out[c*256 + f] = w[f*256 + c]
    int c = i >> 8, f = i & 255;
    wqT[(size_t)c * D_ + f] = f2bf(wq[(size_t)f * D_ + c]);
    wkT[(size_t)c * D_ + f] = f2bf(wk[(size_t)f * D_ + c]);
    wvT[(size_t)c * D_ + f] = f2bf(wv[(size_t)f * D_ + c]);
  }
  if (i < G3D_ * D_) {     // gx = upd @ wih^T  =>  B[k][c] = wih[c*256 + k]: already K-contiguous
    wihB[i] = f2bf(wih[i]);
    whhB[i] = f2bf(whh[i]);
  }
}

// ---------------- slots init: conditioning (B,K,D) -> padded (B,16,D) ----------------
__global__ void init_slots_kernel(const float* __restrict__ cond, float* __restrict__ slots) {
  int idx = blockIdx.x * 256 + threadIdx.x;          // B*16*256
  int d = idx & 255, r = (idx >> 8) & 15, b = idx >> 12;
  slots[idx] = (r < K_) ? cond[((size_t)b * K_ + r) * D_ + d] : 0.f;
}

// ---------------- fused input LN + K/V projection (bf16 WMMA) ------------------------
__global__ __launch_bounds__(128) void ln_project_kernel(
    const float* __restrict__ inp, const float* __restrict__ g, const float* __restrict__ bt,
    const unsigned short* __restrict__ wkT, const unsigned short* __restrict__ wvT,
    unsigned short* __restrict__ kbf, unsigned short* __restrict__ vT) {
  __shared__ __align__(16) unsigned short xs[16 * FD_];
  int tid = threadIdx.x;
  size_t row0 = (size_t)blockIdx.x * 16;
  {   // LayerNorm: 8 lanes per row, 16 rows
    int r = tid >> 3, sub = tid & 7;
    const float* xr = inp + (row0 + r) * FD_;
    float vals[32]; float s = 0.f, ss = 0.f;
    #pragma unroll
    for (int i = 0; i < 32; i++) { float x = xr[sub * 32 + i]; vals[i] = x; s += x; ss += x * x; }
    #pragma unroll
    for (int m = 1; m < 8; m <<= 1) { s += __shfl_xor(s, m, 32); ss += __shfl_xor(ss, m, 32); }
    float mean = s * (1.f / 256.f);
    float inv  = rsqrtf(ss * (1.f / 256.f) - mean * mean + LN_EPS_);
    #pragma unroll
    for (int i = 0; i < 32; i++) {
      int c = sub * 32 + i;
      xs[r * FD_ + c] = f2bf((vals[i] - mean) * inv * g[c] + bt[c]);
    }
  }
  __syncthreads();
  int lane = tid & 31, wave = tid >> 5;
  int col = lane & 15, half = lane >> 4;
  v8f aK[4] = {zero8(), zero8(), zero8(), zero8()};
  v8f aV[4] = {zero8(), zero8(), zero8(), zero8()};
  #pragma unroll
  for (int k0 = 0; k0 < FD_; k0 += 32) {
    v16bf a = load_a_bf16(xs, FD_, k0, lane);
    #pragma unroll
    for (int t = 0; t < 4; t++) {
      int cbase = (wave * 4 + t) * 16;
      v16bf bk = load_b_bf16(wkT + (size_t)cbase * FD_, FD_, k0, lane);
      aK[t] = wmma_bf16(a, bk, aK[t]);
      v16bf bv = load_b_bf16(wvT + (size_t)cbase * FD_, FD_, k0, lane);
      aV[t] = wmma_bf16(a, bv, aV[t]);
    }
  }
  int bI = (int)(row0 / N_), n0 = (int)(row0 % N_);
  #pragma unroll
  for (int t = 0; t < 4; t++) {
    int c = (wave * 4 + t) * 16 + col;
    #pragma unroll
    for (int j = 0; j < 8; j++) {
      int mr = j + 8 * half;
      kbf[(row0 + mr) * D_ + c] = f2bf(aK[t][j]);                        // K: row-major (B*N)xD
      vT[((size_t)bI * D_ + c) * (size_t)N_ + (n0 + mr)] = f2bf(aV[t][j]); // V: transposed B x D x N
    }
  }
}

// ---------------- slot LN + q projection (scaled), padded to 16 rows -----------------
__global__ __launch_bounds__(128) void slot_q_kernel(
    const float* __restrict__ slots, const float* __restrict__ g, const float* __restrict__ bt,
    const unsigned short* __restrict__ wqT, unsigned short* __restrict__ qbf) {
  __shared__ __align__(16) unsigned short xs[16 * D_];
  int b = blockIdx.x, tid = threadIdx.x;
  {
    int r = tid >> 3, sub = tid & 7;
    const float* xr = slots + ((size_t)b * KP + r) * D_;
    float vals[32]; float s = 0.f, ss = 0.f;
    #pragma unroll
    for (int i = 0; i < 32; i++) { float x = xr[sub * 32 + i]; vals[i] = x; s += x; ss += x * x; }
    #pragma unroll
    for (int m = 1; m < 8; m <<= 1) { s += __shfl_xor(s, m, 32); ss += __shfl_xor(ss, m, 32); }
    float mean = s * (1.f / 256.f);
    float inv  = rsqrtf(ss * (1.f / 256.f) - mean * mean + LN_EPS_);
    #pragma unroll
    for (int i = 0; i < 32; i++) {
      int c = sub * 32 + i;
      xs[r * D_ + c] = (r < K_) ? f2bf((vals[i] - mean) * inv * g[c] + bt[c]) : (unsigned short)0;
    }
  }
  __syncthreads();
  int lane = tid & 31, wave = tid >> 5;
  int col = lane & 15, half = lane >> 4;
  v8f acc[4] = {zero8(), zero8(), zero8(), zero8()};
  #pragma unroll
  for (int k0 = 0; k0 < D_; k0 += 32) {
    v16bf a = load_a_bf16(xs, D_, k0, lane);
    #pragma unroll
    for (int t = 0; t < 4; t++) {
      int cbase = (wave * 4 + t) * 16;
      v16bf bb = load_b_bf16(wqT + (size_t)cbase * D_, D_, k0, lane);
      acc[t] = wmma_bf16(a, bb, acc[t]);
    }
  }
  #pragma unroll
  for (int t = 0; t < 4; t++) {
    int c = (wave * 4 + t) * 16 + col;
    #pragma unroll
    for (int j = 0; j < 8; j++) {
      int row = j + 8 * half;
      qbf[((size_t)b * KP + row) * D_ + c] = f2bf(acc[t][j] * SCALE_);   // fold SCALE into q
    }
  }
}

// ---------------- dots = q@k^T (WMMA) + competitive softmax across slots -------------
__global__ __launch_bounds__(128) void dots_softmax_kernel(
    const unsigned short* __restrict__ qbf, const unsigned short* __restrict__ kbf,
    float* __restrict__ attn) {
  int b = blockIdx.x;
  int lane = threadIdx.x & 31, wave = threadIdx.x >> 5;
  int n0 = blockIdx.y * 64 + wave * 16;
  int col = lane & 15, half = lane >> 4;
  v8f acc = zero8();
  const unsigned short* qb = qbf + (size_t)b * KP * D_;
  const unsigned short* kb = kbf + ((size_t)b * N_ + n0) * (size_t)D_;
  #pragma unroll
  for (int k0 = 0; k0 < D_; k0 += 32) {
    v16bf a  = load_a_bf16(qb, D_, k0, lane);
    v16bf bb = load_b_bf16(kb, D_, k0, lane);
    acc = wmma_bf16(a, bb, acc);
  }
  // softmax over slot rows 0..10 per key column; column split across lane pair (l, l^16)
  float mx = -3.0e38f;
  #pragma unroll
  for (int j = 0; j < 8; j++) { int row = j + 8 * half; if (row < K_) mx = fmaxf(mx, acc[j]); }
  mx = fmaxf(mx, __shfl_xor(mx, 16, 32));
  float e[8]; float sum = 0.f;
  #pragma unroll
  for (int j = 0; j < 8; j++) {
    int row = j + 8 * half;
    e[j] = (row < K_) ? __expf(acc[j] - mx) : 0.f;
    sum += e[j];
  }
  sum += __shfl_xor(sum, 16, 32);
  float inv = 1.f / sum;
  int n = n0 + col;
  #pragma unroll
  for (int j = 0; j < 8; j++) {
    int row = j + 8 * half;
    attn[((size_t)b * KP + row) * (size_t)N_ + n] = e[j] * inv;          // padded rows -> 0
  }
}

// ---------------- per-slot attn row sums (deterministic, no atomics) -----------------
__global__ __launch_bounds__(256) void row_sums_kernel(const float* __restrict__ attn,
                                                       float* __restrict__ sums) {
  int br = blockIdx.x, t = threadIdx.x;
  const float* p = attn + (size_t)br * N_;
  float s = 0.f;
  for (int i = t; i < N_; i += 256) s += p[i];
  __shared__ float red[256];
  red[t] = s; __syncthreads();
  for (int st = 128; st > 0; st >>= 1) { if (t < st) red[t] += red[t + st]; __syncthreads(); }
  if (t == 0) sums[br] = red[0];
}

// ---------------- updates = renorm(attn) @ v (WMMA over N=4096) ----------------------
__global__ __launch_bounds__(128) void attn_v_kernel(
    const float* __restrict__ attn, const float* __restrict__ sums,
    const unsigned short* __restrict__ vT, float* __restrict__ upd) {
  int b = blockIdx.x;
  int lane = threadIdx.x & 31, wave = threadIdx.x >> 5;
  int d0 = blockIdx.y * 64 + wave * 16;
  int col = lane & 15, half = lane >> 4;
  int arow = col;
  float denom = sums[b * KP + arow] + EPS_ * (float)N_;   // sum_n(attn + eps)
  float rinv  = 1.f / denom;
  bool zr = (arow >= K_);
  v8f acc = zero8();
  const float* ab = attn + (size_t)b * KP * N_;
  const unsigned short* vb = vT + ((size_t)b * D_ + d0) * (size_t)N_;
  for (int j0 = 0; j0 < N_; j0 += 32) {
    v16bf a  = load_a_f32(ab, N_, j0, lane, EPS_, rinv, zr);
    v16bf bb = load_b_bf16(vb, N_, j0, lane);
    acc = wmma_bf16(a, bb, acc);
  }
  #pragma unroll
  for (int j = 0; j < 8; j++) {
    int row = j + 8 * half;
    upd[((size_t)b * KP + row) * D_ + d0 + col] = acc[j];
  }
}

// ---------------- GRU gate matmuls: gx = upd@wih^T + bih, gh = h@whh^T + bhh ---------
__global__ __launch_bounds__(128) void gru_mm_kernel(
    const float* __restrict__ upd, const float* __restrict__ slots,
    const unsigned short* __restrict__ wihB, const unsigned short* __restrict__ whhB,
    const float* __restrict__ bih, const float* __restrict__ bhh,
    float* __restrict__ gx, float* __restrict__ gh) {
  int b = blockIdx.x;
  int lane = threadIdx.x & 31, wave = threadIdx.x >> 5;
  int col = lane & 15, half = lane >> 4;
  const float* ub = upd   + (size_t)b * KP * D_;
  const float* sb = slots + (size_t)b * KP * D_;
  for (int t = wave; t < G3D_ / 16; t += 4) {
    v8f ax = zero8(), ah = zero8();
    #pragma unroll
    for (int k0 = 0; k0 < D_; k0 += 32) {
      v16bf au = load_a_f32(ub, D_, k0, lane, 0.f, 1.f, false);
      v16bf as = load_a_f32(sb, D_, k0, lane, 0.f, 1.f, false);
      v16bf bi = load_b_bf16(wihB + (size_t)t * 16 * D_, D_, k0, lane);
      v16bf bh = load_b_bf16(whhB + (size_t)t * 16 * D_, D_, k0, lane);
      ax = wmma_bf16(au, bi, ax);
      ah = wmma_bf16(as, bh, ah);
    }
    int c = t * 16 + col;
    #pragma unroll
    for (int j = 0; j < 8; j++) {
      int row = j + 8 * half;
      gx[(size_t)b * KP * G3D_ + (size_t)row * G3D_ + c] = ax[j] + bih[c];
      gh[(size_t)b * KP * G3D_ + (size_t)row * G3D_ + c] = ah[j] + bhh[c];
    }
  }
}

// ---------------- GRU elementwise update ---------------------------------------------
__global__ void gru_update_kernel(const float* __restrict__ gx, const float* __restrict__ gh,
                                  float* __restrict__ slots) {
  int idx = blockIdx.x * 256 + threadIdx.x;   // B*K*D
  int d = idx & 255, t = idx >> 8;
  int b = t / K_, kk = t % K_;
  size_t gbase = ((size_t)b * KP + kk) * G3D_;
  float xr = gx[gbase + d], xz = gx[gbase + 256 + d], xn = gx[gbase + 512 + d];
  float hr = gh[gbase + d], hz = gh[gbase + 256 + d], hn = gh[gbase + 512 + d];
  size_t sidx = ((size_t)b * KP + kk) * D_ + d;
  float h  = slots[sidx];
  float r  = 1.f / (1.f + __expf(-(xr + hr)));
  float z  = 1.f / (1.f + __expf(-(xz + hz)));
  float nt = tanhf(xn + r * hn);
  slots[sidx] = (1.f - z) * nt + z * h;
}

// ---------------- attn_out: drop row padding -----------------------------------------
__global__ void copy_attn_kernel(const float* __restrict__ attn, float* __restrict__ out) {
  size_t idx = (size_t)blockIdx.x * 256 + threadIdx.x;  // B*K*N
  int n = (int)(idx % N_);
  size_t t = idx / N_;
  int kk = (int)(t % K_), b = (int)(t / K_);
  out[idx] = attn[((size_t)b * KP + kk) * (size_t)N_ + n];
}

// ---------------- gumbel head + slot output ------------------------------------------
__global__ __launch_bounds__(256) void finalize_kernel(
    const float* __restrict__ slots, const float* __restrict__ gum_w, const float* __restrict__ gum_b,
    const float* __restrict__ gu, float* __restrict__ out_slots,
    float* __restrict__ out_prob, float* __restrict__ out_hard) {
  int bk = blockIdx.x, b = bk / K_, kk = bk % K_;
  int d = threadIdx.x;
  float s = slots[((size_t)b * KP + kk) * D_ + d];
  out_slots[(size_t)bk * D_ + d] = s;
  __shared__ float r0[256], r1[256];
  r0[d] = s * gum_w[d * 2 + 0];
  r1[d] = s * gum_w[d * 2 + 1];
  __syncthreads();
  for (int st = 128; st > 0; st >>= 1) {
    if (d < st) { r0[d] += r0[d + st]; r1[d] += r1[d + st]; }
    __syncthreads();
  }
  if (d == 0) {
    float l0 = r0[0] + gum_b[0], l1 = r1[0] + gum_b[1];
    float g0 = -__logf(-__logf(gu[bk * 2 + 0]));
    float g1 = -__logf(-__logf(gu[bk * 2 + 1]));
    // TAU = 1; straight-through value == one-hot(argmax); argmax tie -> index 0
    out_hard[bk] = ((l1 + g1) > (l0 + g0)) ? 1.f : 0.f;
    out_prob[bk] = 1.f / (1.f + __expf(l0 - l1));   // softmax(logits)[1]
  }
}

extern "C" void kernel_launch(void* const* d_in, const int* in_sizes, int n_in,
                              void* d_out, int out_size, void* d_ws, size_t ws_size,
                              hipStream_t stream) {
  (void)in_sizes; (void)n_in; (void)out_size; (void)ws_size;
  const float* inputs       = (const float*)d_in[0];
  const float* conditioning = (const float*)d_in[1];
  const float* gumbel_u     = (const float*)d_in[2];
  const float* ln_in_g      = (const float*)d_in[3];
  const float* ln_in_b      = (const float*)d_in[4];
  const float* ln_slot_g    = (const float*)d_in[5];
  const float* ln_slot_b    = (const float*)d_in[6];
  const float* wq           = (const float*)d_in[7];
  const float* wk           = (const float*)d_in[8];
  const float* wv           = (const float*)d_in[9];
  const float* gwih         = (const float*)d_in[10];
  const float* gwhh         = (const float*)d_in[11];
  const float* gbih         = (const float*)d_in[12];
  const float* gbhh         = (const float*)d_in[13];
  const float* gum_w        = (const float*)d_in[14];
  const float* gum_b        = (const float*)d_in[15];

  size_t off = 0;
  auto carve = [&](size_t bytes) -> char* {
    char* p = (char*)d_ws + off;
    off += (bytes + 255) & ~(size_t)255;
    return p;
  };
  unsigned short* kbf  = (unsigned short*)carve((size_t)B_ * N_ * D_ * 2);   // K bf16 (B*N)xD
  unsigned short* vT   = (unsigned short*)carve((size_t)B_ * N_ * D_ * 2);   // V bf16 B x D x N
  unsigned short* qbf  = (unsigned short*)carve((size_t)B_ * KP * D_ * 2);
  unsigned short* wqT  = (unsigned short*)carve((size_t)D_ * D_ * 2);
  unsigned short* wkT  = (unsigned short*)carve((size_t)D_ * D_ * 2);
  unsigned short* wvTb = (unsigned short*)carve((size_t)D_ * D_ * 2);
  unsigned short* wihB = (unsigned short*)carve((size_t)G3D_ * D_ * 2);
  unsigned short* whhB = (unsigned short*)carve((size_t)G3D_ * D_ * 2);
  float* slots = (float*)carve((size_t)B_ * KP * D_ * 4);
  float* attn  = (float*)carve((size_t)B_ * KP * N_ * 4);
  float* sums  = (float*)carve((size_t)B_ * KP * 4);
  float* upd   = (float*)carve((size_t)B_ * KP * D_ * 4);
  float* gx    = (float*)carve((size_t)B_ * KP * G3D_ * 4);
  float* gh    = (float*)carve((size_t)B_ * KP * G3D_ * 4);

  float* out_slots = (float*)d_out;                     // (B,K,D)
  float* out_attn  = out_slots + (size_t)B_ * K_ * D_;  // (B,K,N)
  float* out_prob  = out_attn + (size_t)B_ * K_ * N_;   // (B,K)
  float* out_hard  = out_prob + (size_t)B_ * K_;        // (B,K)

  prep_weights_kernel<<<(G3D_ * D_) / 256, 256, 0, stream>>>(
      wq, wk, wv, gwih, gwhh, wqT, wkT, wvTb, wihB, whhB);
  init_slots_kernel<<<(B_ * KP * D_) / 256, 256, 0, stream>>>(conditioning, slots);
  ln_project_kernel<<<(B_ * N_) / 16, 128, 0, stream>>>(
      inputs, ln_in_g, ln_in_b, wkT, wvTb, kbf, vT);

  for (int it = 0; it < ITERS_; ++it) {
    slot_q_kernel<<<B_, 128, 0, stream>>>(slots, ln_slot_g, ln_slot_b, wqT, qbf);
    dots_softmax_kernel<<<dim3(B_, N_ / 64), 128, 0, stream>>>(qbf, kbf, attn);
    row_sums_kernel<<<B_ * KP, 256, 0, stream>>>(attn, sums);
    attn_v_kernel<<<dim3(B_, 4), 128, 0, stream>>>(attn, sums, vT, upd);
    gru_mm_kernel<<<B_, 128, 0, stream>>>(upd, slots, wihB, whhB, gbih, gbhh, gx, gh);
    gru_update_kernel<<<(B_ * K_ * D_) / 256, 256, 0, stream>>>(gx, gh, slots);
  }

  copy_attn_kernel<<<(B_ * K_ * N_) / 256, 256, 0, stream>>>(attn, out_attn);
  finalize_kernel<<<B_ * K_, 256, 0, stream>>>(
      slots, gum_w, gum_b, gumbel_u, out_slots, out_prob, out_hard);
}
